// SectionAttention_23974507446660
// MI455X (gfx1250) — matrix operand order repaired and checked by reading, not compiled, not run on hardware
//
#include <hip/hip_runtime.h>
#include <hip/hip_bf16.h>

// Problem dims (fixed by reference)
#define NB 8
#define NL 256
#define NG 8
#define NS 64
#define NE 1024
#define NH 16
#define NHD 64
#define NS2 66
#define QSCALE 0.125f   // HD^-0.5, 64^-0.5

#define KSTEP  32
#define LDSPAD 8
#define LDW    (KSTEP + LDSPAD)   // 40 bf16 per staged row (80 B, keeps 16B align)

typedef __bf16 bf16_t;
typedef __attribute__((ext_vector_type(16))) __bf16 bf16x16;
typedef __attribute__((ext_vector_type(8)))  __bf16 bf16x8;
typedef __attribute__((ext_vector_type(4)))  __bf16 bf16x4;
typedef __attribute__((ext_vector_type(8)))  float  f32x8;
typedef __attribute__((ext_vector_type(4)))  float  f32x4;
typedef int vint4 __attribute__((vector_size(16)));   // matches builtin param

#define AS1 __attribute__((address_space(1)))
#define AS3 __attribute__((address_space(3)))

union FragBF16 { bf16x16 v; bf16x8 h[2]; };

// ---------------------------------------------------------------------------
// async 16-byte global -> LDS copy (ASYNCcnt-tracked), builtin if present,
// inline-asm fallback otherwise.
// ---------------------------------------------------------------------------
__device__ __forceinline__ void async_cp16(const bf16_t* g, bf16_t* l) {
#if __has_builtin(__builtin_amdgcn_global_load_async_to_lds_b128)
    __builtin_amdgcn_global_load_async_to_lds_b128(
        (AS1 vint4*)(unsigned long long)(const void*)g,
        (AS3 vint4*)(unsigned)(unsigned long long)(const void*)l,
        0, 0);
#else
    unsigned loff = (unsigned)(unsigned long long)(const void*)l;
    asm volatile("global_load_async_to_lds_b128 %0, %1, off"
                 :: "v"(loff), "v"(g) : "memory");
#endif
}

__device__ __forceinline__ void async_wait4() {
#if __has_builtin(__builtin_amdgcn_s_wait_asynccnt)
    __builtin_amdgcn_s_wait_asynccnt(4);
#else
    asm volatile("s_wait_asynccnt 0x4" ::: "memory");
#endif
}

__device__ __forceinline__ void async_wait0() {
#if __has_builtin(__builtin_amdgcn_s_wait_asynccnt)
    __builtin_amdgcn_s_wait_asynccnt(0);
#else
    asm volatile("s_wait_asynccnt 0x0" ::: "memory");
#endif
}

// ---------------------------------------------------------------------------
// 1) per-token best-section argmax over attn_scores[:, :, :G]
// ---------------------------------------------------------------------------
__global__ void best_section_kernel(const float* __restrict__ scores,
                                    int* __restrict__ best) {
    int i = blockIdx.x * blockDim.x + threadIdx.x;   // i = b*NL + l
    if (i >= NB * NL) return;
    const float* p = scores + (size_t)i * (NG + 2);
    int   bi = 0;
    float bv = p[0];
    #pragma unroll
    for (int g = 1; g < NG; ++g) {
        float v = p[g];
        if (v > bv) { bv = v; bi = g; }   // first-max tie rule like jnp.argmax
    }
    best[i] = bi;
}

// ---------------------------------------------------------------------------
// 2) f32 -> bf16 conversion (vectorized 4/thread)
// ---------------------------------------------------------------------------
__global__ void cvt_f32_to_bf16_kernel(const float* __restrict__ in,
                                       bf16_t* __restrict__ out, int n4) {
    int i = blockIdx.x * blockDim.x + threadIdx.x;
    if (i >= n4) return;
    f32x4 v = ((const f32x4*)in)[i];
    bf16x4 o;
    o.x = (bf16_t)v.x; o.y = (bf16_t)v.y; o.z = (bf16_t)v.z; o.w = (bf16_t)v.w;
    ((bf16x4*)out)[i] = o;
}

// ---------------------------------------------------------------------------
// 3) WMMA GEMM:  C[M,N] = (A[M,K] x B[N,K]^T + bias[N]) * scale
//    A,B bf16 row-major (NT GEMM, K-contiguous both sides).
//    Block: 256 thr = 8 waves; wave tile 64x32 (4x2 of 16x16); block 128x128.
//    Per K-step: 128x32 A and B panels staged into double-buffered LDS via
//    GLOBAL_LOAD_ASYNC_TO_LDS_B128, fragments read back with ds_load_b128.
// ---------------------------------------------------------------------------
__global__ __launch_bounds__(256)
void gemm_bf16_wmma_kernel(const bf16_t* __restrict__ A,
                           const bf16_t* __restrict__ Bw,
                           const float*  __restrict__ bias,
                           bf16_t* __restrict__ outB,
                           float*  __restrict__ outF,
                           int M, int N, int K, float scale) {
    __shared__ bf16_t As[2][128 * LDW];
    __shared__ bf16_t Bs[2][128 * LDW];

    const int lane     = threadIdx.x & 31;
    const int wave     = threadIdx.x >> 5;   // 0..7
    const int waveM    = wave & 1;           // 2 waves along M
    const int waveN    = wave >> 1;          // 4 waves along N
    const int laneRow  = lane & 15;
    const int laneHalf = lane >> 4;

    const int blockM = blockIdx.y * 128;
    const int blockN = blockIdx.x * 128;

    // staging map: 512 16B-chunks per panel; chunk c -> row c/4, col (c%4)*8
    const int c0   = threadIdx.x;            // chunks c0 and c0+256
    const int row0 = c0 >> 2,        col0 = (c0 & 3) * 8;
    const int row1 = (c0 + 256) >> 2, col1 = ((c0 + 256) & 3) * 8;

    const bf16_t* aG0 = A  + (size_t)(blockM + row0) * K + col0;
    const bf16_t* aG1 = A  + (size_t)(blockM + row1) * K + col1;
    const bf16_t* bG0 = Bw + (size_t)(blockN + row0) * K + col0;
    const bf16_t* bG1 = Bw + (size_t)(blockN + row1) * K + col1;
    const int aL0 = row0 * LDW + col0, aL1 = row1 * LDW + col1;

    f32x8 zero = {0.f, 0.f, 0.f, 0.f, 0.f, 0.f, 0.f, 0.f};
    f32x8 acc[4][2];
    #pragma unroll
    for (int mi = 0; mi < 4; ++mi)
        #pragma unroll
        for (int ni = 0; ni < 2; ++ni) acc[mi][ni] = zero;

    // fragment read pointers inside the staged panels
    // A (16-bit 16x32 layout): lane<16 -> K {0..7,16..23}; lane>=16 -> {8..15,24..31}
    const int aFra = (waveM * 64 + laneRow) * LDW + laneHalf * 8;
    // B (16-bit 32x16 layout): lane<16 -> K {0..7,16..23}; lane>=16 -> {8..15,24..31}
    const int bFra = (waveN * 32 + laneRow) * LDW + laneHalf * 8;

    // prologue: stage K-step 0 into buffer 0
    async_cp16(aG0, &As[0][aL0]);
    async_cp16(aG1, &As[0][aL1]);
    async_cp16(bG0, &Bs[0][aL0]);
    async_cp16(bG1, &Bs[0][aL1]);

    int st = 0;
    for (int k0 = 0; k0 < K; k0 += KSTEP) {
        if (k0 + KSTEP < K) {
            const int kn = k0 + KSTEP;
            async_cp16(aG0 + kn, &As[st ^ 1][aL0]);
            async_cp16(aG1 + kn, &As[st ^ 1][aL1]);
            async_cp16(bG0 + kn, &Bs[st ^ 1][aL0]);
            async_cp16(bG1 + kn, &Bs[st ^ 1][aL1]);
            async_wait4();           // current stage landed; prefetch in flight
        } else {
            async_wait0();
        }
        __syncthreads();

        FragBF16 afr[4];
        #pragma unroll
        for (int t = 0; t < 4; ++t) {
            const bf16_t* p = &As[st][aFra + t * 16 * LDW];
            afr[t].h[0] = *(const bf16x8*)(p);
            afr[t].h[1] = *(const bf16x8*)(p + 16);
        }
        FragBF16 bfr[2];
        #pragma unroll
        for (int t = 0; t < 2; ++t) {
            const bf16_t* p = &Bs[st][bFra + t * 16 * LDW];
            bfr[t].h[0] = *(const bf16x8*)(p);
            bfr[t].h[1] = *(const bf16x8*)(p + 16);
        }

        #pragma unroll
        for (int mi = 0; mi < 4; ++mi)
            #pragma unroll
            for (int ni = 0; ni < 2; ++ni)
                acc[mi][ni] = __builtin_amdgcn_wmma_f32_16x16x32_bf16(
                    false, afr[mi].v, false, bfr[ni].v,
                    (short)0, acc[mi][ni], false, false);

        __syncthreads();             // frags consumed before buffer reuse
        st ^= 1;
    }

    // Epilogue. C/D layout: lane -> N = lane%16; VGPR r -> M = r + (lane/16)*8
    #pragma unroll
    for (int mi = 0; mi < 4; ++mi) {
        #pragma unroll
        for (int ni = 0; ni < 2; ++ni) {
            int gn = blockN + waveN * 32 + ni * 16 + laneRow;
            float bv = bias ? bias[gn] : 0.0f;
            #pragma unroll
            for (int r = 0; r < 8; ++r) {
                int gm = blockM + waveM * 64 + mi * 16 + r + laneHalf * 8;
                float v = (acc[mi][ni][r] + bv) * scale;
                if (outB) outB[(size_t)gm * N + gn] = (bf16_t)v;
                else      outF[(size_t)gm * N + gn] = v;
            }
        }
    }
}

// ---------------------------------------------------------------------------
// 4) routed attention over S+2=66 slots, per (b,l): all 16 heads in one block.
// ---------------------------------------------------------------------------
__global__ __launch_bounds__(256)
void attention_kernel(const bf16_t* __restrict__ Q,    // [NL*NB, NE], row l*NB+b
                      const bf16_t* __restrict__ Kp,   // [NB*NG*NS, NE]
                      const bf16_t* __restrict__ Vp,   // [NB*NG*NS, NE]
                      const unsigned char* __restrict__ mask, // [NB*NG*NS] bool
                      const float* __restrict__ bias_k,       // [NE]
                      const float* __restrict__ bias_v,       // [NE]
                      const int*   __restrict__ best,         // [NB*NL]
                      bf16_t* __restrict__ attnOut,           // [NL*NB, NE]
                      float*  __restrict__ wOut) {            // [NB*NL, 66]
    __shared__ float sw[NH][NS2 + 2];

    const int bl = blockIdx.x;       // b*NL + l
    const int b  = bl / NL;
    const int l  = bl % NL;
    const int t  = threadIdx.x;
    const int h  = t >> 4;           // head 0..15
    const int j  = t & 15;           // lane within head group

    const int g        = best[bl];
    const size_t krow0 = (size_t)((b * NG + g) * NS) * NE;
    const int    eoff  = h * NHD + j * 4;

    bf16x4 qv = *(const bf16x4*)(Q + (size_t)(l * NB + b) * NE + eoff);
    float q0 = (float)qv.x, q1 = (float)qv.y, q2 = (float)qv.z, q3 = (float)qv.w;

    const unsigned char* mrow = mask + (size_t)(b * NG + g) * NS;

    // ---- scores over real slots ----
    for (int s = 0; s < NS; ++s) {
        bf16x4 kv = *(const bf16x4*)(Kp + krow0 + (size_t)s * NE + eoff);
        float p = q0 * (float)kv.x + q1 * (float)kv.y +
                  q2 * (float)kv.z + q3 * (float)kv.w;
        #pragma unroll
        for (int o = 8; o >= 1; o >>= 1) p += __shfl_xor(p, o, 16);
        if (j == 0) sw[h][s] = mrow[s] ? -__builtin_inff() : p;
    }
    // ---- bias_k slot (s=64) and zero slot (s=65), never masked ----
    {
        const float* bk = bias_k + eoff;
        float p = q0 * bk[0] + q1 * bk[1] + q2 * bk[2] + q3 * bk[3];
        #pragma unroll
        for (int o = 8; o >= 1; o >>= 1) p += __shfl_xor(p, o, 16);
        if (j == 0) { sw[h][64] = p; sw[h][65] = 0.0f; }
    }

    // ---- softmax over 66 slots (per head group) ----
    float m = -__builtin_inff();
    #pragma unroll
    for (int i = 0; i < 5; ++i) {
        int s = j + 16 * i;
        if (s < NS2) m = fmaxf(m, sw[h][s]);
    }
    #pragma unroll
    for (int o = 8; o >= 1; o >>= 1) m = fmaxf(m, __shfl_xor(m, o, 16));
    float sum = 0.0f;
    #pragma unroll
    for (int i = 0; i < 5; ++i) {
        int s = j + 16 * i;
        if (s < NS2) { float e = __expf(sw[h][s] - m); sw[h][s] = e; sum += e; }
    }
    #pragma unroll
    for (int o = 8; o >= 1; o >>= 1) sum += __shfl_xor(sum, o, 16);
    float inv = 1.0f / sum;
    #pragma unroll
    for (int i = 0; i < 5; ++i) {
        int s = j + 16 * i;
        if (s < NS2) sw[h][s] *= inv;
    }

    // ---- weighted V ----
    float a0 = 0.f, a1 = 0.f, a2 = 0.f, a3 = 0.f;
    for (int s = 0; s < NS; ++s) {
        float w = sw[h][s];
        bf16x4 vv = *(const bf16x4*)(Vp + krow0 + (size_t)s * NE + eoff);
        a0 += w * (float)vv.x; a1 += w * (float)vv.y;
        a2 += w * (float)vv.z; a3 += w * (float)vv.w;
    }
    {
        float w = sw[h][64];
        const float* bv = bias_v + eoff;
        a0 += w * bv[0]; a1 += w * bv[1]; a2 += w * bv[2]; a3 += w * bv[3];
        // slot 65 contributes zero
    }
    bf16x4 ov;
    ov.x = (bf16_t)a0; ov.y = (bf16_t)a1; ov.z = (bf16_t)a2; ov.w = (bf16_t)a3;
    *(bf16x4*)(attnOut + (size_t)(l * NB + b) * NE + eoff) = ov;

    // ---- attn_weights = mean over heads ----
    __syncthreads();
    if (t < NS2) {
        float s = 0.0f;
        #pragma unroll
        for (int hh = 0; hh < NH; ++hh) s += sw[hh][t];
        wOut[(size_t)bl * NS2 + t] = s * (1.0f / NH);
    }
}

// ---------------------------------------------------------------------------
// launch
// ---------------------------------------------------------------------------
extern "C" void kernel_launch(void* const* d_in, const int* in_sizes, int n_in,
                              void* d_out, int out_size, void* d_ws, size_t ws_size,
                              hipStream_t stream) {
    const float* query       = (const float*)d_in[0];   // [NL,NB,NE]
    const float* sections    = (const float*)d_in[1];   // [NB,NG,NS,NE]
    const unsigned char* msk = (const unsigned char*)d_in[2]; // [NB,NG,NS] bool
    const float* attn_scores = (const float*)d_in[3];   // [NB,NL,NG+2]
    const float* Wq = (const float*)d_in[4];
    const float* bq = (const float*)d_in[5];
    const float* Wk = (const float*)d_in[6];
    const float* bk = (const float*)d_in[7];
    const float* Wv = (const float*)d_in[8];
    const float* bv = (const float*)d_in[9];
    const float* Wo = (const float*)d_in[10];
    const float* bo = (const float*)d_in[11];
    const float* bias_k = (const float*)d_in[12];       // [NE]
    const float* bias_v = (const float*)d_in[13];       // [NE]

    const size_t MQ = (size_t)NL * NB;       // 2048 rows (l,b)
    const size_t MS = (size_t)NB * NG * NS;  // 4096 section rows

    // workspace carve-up (~44 MB)
    char* ws = (char*)d_ws;
    size_t off = 0;
    auto carve = [&](size_t bytes) -> void* {
        void* p = ws + off;
        off = (off + bytes + 255) & ~(size_t)255;
        return p;
    };
    int*    best  = (int*)   carve(MQ * sizeof(int));
    bf16_t* qx    = (bf16_t*)carve(MQ * NE * 2);   // query bf16
    bf16_t* sx    = (bf16_t*)carve(MS * NE * 2);   // sections bf16
    bf16_t* wq    = (bf16_t*)carve((size_t)NE * NE * 2);
    bf16_t* wk    = (bf16_t*)carve((size_t)NE * NE * 2);
    bf16_t* wv    = (bf16_t*)carve((size_t)NE * NE * 2);
    bf16_t* wo    = (bf16_t*)carve((size_t)NE * NE * 2);
    bf16_t* qproj = (bf16_t*)carve(MQ * NE * 2);   // (XWq^T+bq)*scale, bf16
    bf16_t* kproj = (bf16_t*)carve(MS * NE * 2);   // section K
    bf16_t* vproj = (bf16_t*)carve(MS * NE * 2);   // section V
    bf16_t* abuf  = (bf16_t*)carve(MQ * NE * 2);   // attention head-concat

    float* out_main = (float*)d_out;                         // [NL,NB,NE]
    float* out_w    = (float*)d_out + MQ * NE;               // [NB,NL,66]

    // 1) routing
    best_section_kernel<<<(NB * NL + 255) / 256, 256, 0, stream>>>(attn_scores, best);

    // 2) bf16 conversions
    {
        int n4;
        n4 = (int)(MQ * NE / 4);
        cvt_f32_to_bf16_kernel<<<(n4 + 255) / 256, 256, 0, stream>>>(query, qx, n4);
        n4 = (int)(MS * NE / 4);
        cvt_f32_to_bf16_kernel<<<(n4 + 255) / 256, 256, 0, stream>>>(sections, sx, n4);
        n4 = (int)((size_t)NE * NE / 4);
        cvt_f32_to_bf16_kernel<<<(n4 + 255) / 256, 256, 0, stream>>>(Wq, wq, n4);
        cvt_f32_to_bf16_kernel<<<(n4 + 255) / 256, 256, 0, stream>>>(Wk, wk, n4);
        cvt_f32_to_bf16_kernel<<<(n4 + 255) / 256, 256, 0, stream>>>(Wv, wv, n4);
        cvt_f32_to_bf16_kernel<<<(n4 + 255) / 256, 256, 0, stream>>>(Wo, wo, n4);
    }

    // 3) projections (all NT GEMMs, WMMA bf16, f32 accumulate)
    {
        dim3 gq(NE / 128, (unsigned)(MQ / 128));
        gemm_bf16_wmma_kernel<<<gq, 256, 0, stream>>>(
            qx, wq, bq, qproj, nullptr, (int)MQ, NE, NE, QSCALE);
        dim3 gs(NE / 128, (unsigned)(MS / 128));
        gemm_bf16_wmma_kernel<<<gs, 256, 0, stream>>>(
            sx, wk, bk, kproj, nullptr, (int)MS, NE, NE, 1.0f);
        gemm_bf16_wmma_kernel<<<gs, 256, 0, stream>>>(
            sx, wv, bv, vproj, nullptr, (int)MS, NE, NE, 1.0f);
    }

    // 4) routed attention (+ attn_weights head-mean)
    attention_kernel<<<NB * NL, 256, 0, stream>>>(
        qproj, kproj, vproj, msk, bias_k, bias_v, best, abuf, out_w);

    // 5) output projection -> f32 d_out
    {
        dim3 go(NE / 128, (unsigned)(MQ / 128));
        gemm_bf16_wmma_kernel<<<go, 256, 0, stream>>>(
            abuf, wo, bo, nullptr, out_main, (int)MQ, NE, NE, 1.0f);
    }
}
